// Basic_block_15126874817248
// MI455X (gfx1250) — compile-verified, matrix operand order checked by reading, not run on hardware
//
#include <hip/hip_runtime.h>
#include <hip/hip_bf16.h>
#include <math.h>

typedef __attribute__((ext_vector_type(16))) __bf16 v16bf;
typedef __attribute__((ext_vector_type(8)))  float  v8f;
typedef unsigned int u32x4 __attribute__((ext_vector_type(4)));
typedef int          i32x8 __attribute__((ext_vector_type(8)));
typedef int          i32x4 __attribute__((ext_vector_type(4)));

union Frag { v16bf v; unsigned u[8]; };

__device__ __forceinline__ __bf16 f2bf(float f) {
  unsigned u; __builtin_memcpy(&u, &f, 4);
  unsigned r = u + 0x7FFFu + ((u >> 16) & 1u);
  unsigned short h = (unsigned short)(r >> 16);
  __bf16 b; __builtin_memcpy(&b, &h, 2);
  return b;
}
__device__ __forceinline__ float bf2f(__bf16 b) {
  unsigned short h; __builtin_memcpy(&h, &b, 2);
  unsigned u = ((unsigned)h) << 16;
  float f; __builtin_memcpy(&f, &u, 4);
  return f;
}

// ------------------------------------------------------------ TDM 2-D tile load
// Builds a D# descriptor (group0 + group1, groups 2/3 zero for a 2-D tensor)
// and issues TENSOR_LOAD_TO_LDS (6-arg clang-23 form). data_size = 2 bytes (bf16).
// tile: tile_d0 elems per row, tile_d1 rows; row stride = stride_elems.
__device__ __forceinline__ void tdm_load_2d(unsigned lds_off, const void* gptr,
                                            int tile_d0, int tile_d1,
                                            unsigned tensor_d0, unsigned tensor_d1,
                                            unsigned stride_elems) {
  unsigned long long ga = (unsigned long long)(uintptr_t)gptr;
  u32x4 g0;
  g0[0] = 1u;                                        // count=1 (valid), user mode
  g0[1] = lds_off;                                   // lds_addr (bytes)
  g0[2] = (unsigned)(ga & 0xFFFFFFFFu);              // global_addr[31:0]
  g0[3] = (unsigned)((ga >> 32) & 0x1FFFFFFu) | (2u << 30);  // addr[56:32] | type=2
  i32x8 g1;
  g1[0] = (int)(1u << 16);                           // data_size=1 (2 bytes), mask=0
  g1[1] = (int)(tensor_d0 << 16);                    // tensor_dim0[15:0] @ bits 63:48
  g1[2] = (int)((tensor_d0 >> 16) | (tensor_d1 << 16));      // dim0 hi | dim1 lo
  g1[3] = (int)((tensor_d1 >> 16) | ((unsigned)tile_d0 << 16)); // dim1 hi | tile_dim0
  g1[4] = tile_d1 & 0xFFFF;                          // tile_dim1 (tile_dim2=0)
  g1[5] = (int)stride_elems;                         // tensor_dim0_stride[31:0]
  g1[6] = 0;                                         // stride hi, dim1_stride lo
  g1[7] = 0;
  i32x4 z4 = {0, 0, 0, 0};
  i32x8 z8 = {0, 0, 0, 0, 0, 0, 0, 0};
  __builtin_amdgcn_tensor_load_to_lds(g0, g1, z4, z4, z8, 0);
}

// ---------------------------------------------------------------- constants
#define CB   256
#define NWIN 1024        // B * 16 * 16
#define NTOK 65536       // B * 128 * 128

// ---------------------------------------------------------------- f32->bf16
__global__ __launch_bounds__(256) void cvt_bf16_kernel(const float* __restrict__ in,
                                                       __bf16* __restrict__ out, int n) {
  int i = blockIdx.x * 256 + threadIdx.x;
  if (i < n) out[i] = f2bf(in[i]);
}

// ---------------------------------------------------------------- LayerNorm
// one wave32 per token (C=256 -> 8 elems/lane). permute=1: write window order.
__global__ __launch_bounds__(256) void ln_kernel(const float* __restrict__ x,
                                                 const float* __restrict__ g,
                                                 const float* __restrict__ b,
                                                 __bf16* __restrict__ outb,
                                                 float* __restrict__ outf,
                                                 int permute) {
  int tok  = blockIdx.x * 8 + (threadIdx.x >> 5);
  int lane = threadIdx.x & 31;
  const float* xr = x + (size_t)tok * CB;
  float v[8]; float s = 0.f;
#pragma unroll
  for (int i = 0; i < 8; ++i) { v[i] = xr[i * 32 + lane]; s += v[i]; }
#pragma unroll
  for (int o = 16; o > 0; o >>= 1) s += __shfl_xor(s, o, 32);
  float mean = s * (1.f / 256.f);
  float vs = 0.f;
#pragma unroll
  for (int i = 0; i < 8; ++i) { float d = v[i] - mean; vs += d * d; }
#pragma unroll
  for (int o = 16; o > 0; o >>= 1) vs += __shfl_xor(vs, o, 32);
  float rstd = rsqrtf(vs * (1.f / 256.f) + 1e-5f);

  size_t drow = (size_t)tok;
  if (permute) {
    int bb = tok >> 14, rem = tok & 16383;
    int yy = rem >> 7, xx = rem & 127;
    int hb = yy >> 3, ih = yy & 7, wb = xx >> 3, iw = xx & 7;
    drow = (size_t)(((bb * 256 + hb * 16 + wb) << 6) + ih * 8 + iw);
  }
#pragma unroll
  for (int i = 0; i < 8; ++i) {
    int c = i * 32 + lane;
    float y = (v[i] - mean) * rstd * g[c] + b[c];
    outb[drow * CB + c] = f2bf(y);
    if (outf) outf[(size_t)tok * CB + c] = y;
  }
}

// ---------------------------------------------------------------- GEMM (bf16 WMMA)
// C[M,N] = A[M,K] * W[N,K]^T, macro tile 128x64, wave tile 32x32 (4 WMMA/k-step).
// Tiles staged to LDS by the Tensor Data Mover (wave 0 issues, TENSORcnt wait).
// mode 0: out bf16 = acc + bias                        (QKV)
// mode 1: out f32[perm(row)] = add0[perm(row)] + acc   (proj + residual, un-permute)
// mode 2: out bf16 = relu(acc + bias)                  (up)
// mode 3: out f32 = add0 + add1 + acc + bias           (down + residuals -> d_out)
__global__ __launch_bounds__(256) void gemm_bf16_kernel(
    const __bf16* __restrict__ A, int lda,
    const __bf16* __restrict__ W,
    const float* __restrict__ bias,
    int M, int N, int K, int mode,
    __bf16* __restrict__ outb, int ldo,
    float* __restrict__ outf,
    const float* __restrict__ add0,
    const float* __restrict__ add1) {
  __shared__ __align__(16) __bf16 aT[128 * 32];
  __shared__ __align__(16) __bf16 bT[64 * 32];

  int m0 = blockIdx.y * 128, n0 = blockIdx.x * 64;
  int t = threadIdx.x, lane = t & 31, w = t >> 5;
  int half = lane >> 4, lrow = lane & 15;
  int mwave = (w & 3) * 32, nwave = (w >> 2) * 32;

  v8f acc[2][2] = {{{}, {}}, {{}, {}}};

  for (int k0 = 0; k0 < K; k0 += 32) {
    if (w == 0) {                        // TDM: one issuing wave per WG
      tdm_load_2d((unsigned)(uintptr_t)aT, A + (size_t)m0 * lda + k0,
                  32, 128, (unsigned)K, (unsigned)M, (unsigned)lda);
      tdm_load_2d((unsigned)(uintptr_t)bT, W + (size_t)n0 * K + k0,
                  32, 64, (unsigned)K, (unsigned)N, (unsigned)K);
      __builtin_amdgcn_s_wait_tensorcnt(0);
    }
    __syncthreads();

    Frag a0, a1, b0, b1;
    const unsigned* aw = (const unsigned*)aT;
    const unsigned* bw = (const unsigned*)bT;
#pragma unroll
    for (int gg = 0; gg < 8; ++gg) {
      int kk = ((gg & 4) ? 16 : 0) + half * 8 + (gg & 3) * 2;
      a0.u[gg] = aw[((mwave + lrow) * 32 + kk) >> 1];
      a1.u[gg] = aw[((mwave + 16 + lrow) * 32 + kk) >> 1];
      b0.u[gg] = bw[((nwave + lrow) * 32 + kk) >> 1];
      b1.u[gg] = bw[((nwave + 16 + lrow) * 32 + kk) >> 1];
    }
    acc[0][0] = __builtin_amdgcn_wmma_f32_16x16x32_bf16(false, a0.v, false, b0.v, (short)0, acc[0][0], false, false);
    acc[0][1] = __builtin_amdgcn_wmma_f32_16x16x32_bf16(false, a0.v, false, b1.v, (short)0, acc[0][1], false, false);
    acc[1][0] = __builtin_amdgcn_wmma_f32_16x16x32_bf16(false, a1.v, false, b0.v, (short)0, acc[1][0], false, false);
    acc[1][1] = __builtin_amdgcn_wmma_f32_16x16x32_bf16(false, a1.v, false, b1.v, (short)0, acc[1][1], false, false);
    __syncthreads();
  }

#pragma unroll
  for (int bi = 0; bi < 2; ++bi) {
    int col = n0 + nwave + bi * 16 + lrow;
    float bia = (mode == 1) ? 0.f : bias[col];
#pragma unroll
    for (int ai = 0; ai < 2; ++ai) {
#pragma unroll
      for (int r = 0; r < 8; ++r) {
        int row = m0 + mwave + ai * 16 + r + 8 * half;
        float v0 = acc[ai][bi][r] + bia;
        if (mode == 0) {
          outb[(size_t)row * ldo + col] = f2bf(v0);
        } else if (mode == 2) {
          outb[(size_t)row * ldo + col] = f2bf(fmaxf(v0, 0.f));
        } else if (mode == 1) {
          int win = row >> 6, n = row & 63;
          int bb = win >> 8, hb = (win >> 4) & 15, wb = win & 15;
          int ih = n >> 3, iw = n & 7;
          int tok = bb * 16384 + (hb * 8 + ih) * 128 + (wb * 8 + iw);
          size_t o0 = (size_t)tok * CB + col;
          outf[o0] = add0[o0] + v0;
        } else {
          size_t o0 = (size_t)row * CB + col;
          outf[o0] = add0[o0] + add1[o0] + v0;
        }
      }
    }
  }
}

// ---------------------------------------------------------------- attention
// one block per (window, head): S = qk^T*scale + rpb, softmax, O = P*V (WMMA).
// q/k/v tiles (64x32, row stride 768) staged to LDS by the TDM.
__global__ __launch_bounds__(256) void attn_kernel(const __bf16* __restrict__ qkv,
                                                   const float* __restrict__ rpb,
                                                   const int* __restrict__ rel,
                                                   __bf16* __restrict__ out) {
  int win = blockIdx.x >> 3, h = blockIdx.x & 7;
  __shared__ __align__(16) __bf16 qs[64 * 32];
  __shared__ __align__(16) __bf16 ks[64 * 32];
  __shared__ __align__(16) __bf16 vs[64 * 32];
  __shared__ float  S[64 * 64];
  __shared__ __bf16 P[64 * 64];

  int t = threadIdx.x, lane = t & 31, w = t >> 5;
  int half = lane >> 4, lrow = lane & 15;

  if (w == 0) {
    const __bf16* base = qkv + (size_t)win * 64 * 768 + h * 32;
    tdm_load_2d((unsigned)(uintptr_t)qs, base,       32, 64, 768u, (unsigned)(NWIN * 64), 768u);
    tdm_load_2d((unsigned)(uintptr_t)ks, base + 256, 32, 64, 768u, (unsigned)(NWIN * 64), 768u);
    tdm_load_2d((unsigned)(uintptr_t)vs, base + 512, 32, 64, 768u, (unsigned)(NWIN * 64), 768u);
    __builtin_amdgcn_s_wait_tensorcnt(0);
  }
  __syncthreads();

  const float scale = 0.17677669529663687f;  // 1/sqrt(HD=32)
  const unsigned* qw = (const unsigned*)qs;
  const unsigned* kw = (const unsigned*)ks;
#pragma unroll
  for (int i = 0; i < 2; ++i) {              // wave does 2 of 16 S-tiles
    int ti = w * 2 + i, mi = ti >> 2, ni = ti & 3;
    Frag af, bf;
#pragma unroll
    for (int gg = 0; gg < 8; ++gg) {
      int kk = ((gg & 4) ? 16 : 0) + half * 8 + (gg & 3) * 2;
      af.u[gg] = qw[((mi * 16 + lrow) * 32 + kk) >> 1];
      bf.u[gg] = kw[((ni * 16 + lrow) * 32 + kk) >> 1];
    }
    v8f acc = {};
    acc = __builtin_amdgcn_wmma_f32_16x16x32_bf16(false, af.v, false, bf.v, (short)0, acc, false, false);
    int col = ni * 16 + lrow;
#pragma unroll
    for (int r = 0; r < 8; ++r) {
      int row = mi * 16 + r + 8 * half;
      float bia = rpb[rel[row * 64 + col] * 8 + h];
      S[row * 64 + col] = acc[r] * scale + bia;
    }
  }
  __syncthreads();

  if (t < 64) {                               // row softmax
    float mx = -1e30f;
    for (int j = 0; j < 64; ++j) mx = fmaxf(mx, S[t * 64 + j]);
    float sum = 0.f;
    for (int j = 0; j < 64; ++j) { float e = __expf(S[t * 64 + j] - mx); S[t * 64 + j] = e; sum += e; }
    float inv = 1.f / sum;
    for (int j = 0; j < 64; ++j) P[t * 64 + j] = f2bf(S[t * 64 + j] * inv);
  }
  __syncthreads();

  {                                           // O = P @ V, wave -> one 16x16 tile
    int mi = w & 3, ni = w >> 2;
    const unsigned* pw = (const unsigned*)P;
    const unsigned short* vu = (const unsigned short*)vs;
    v8f acc = {};
#pragma unroll
    for (int ks2 = 0; ks2 < 2; ++ks2) {
      Frag af, bf;
#pragma unroll
      for (int gg = 0; gg < 8; ++gg) {
        int kk = ((gg & 4) ? 16 : 0) + half * 8 + (gg & 3) * 2 + ks2 * 32;
        af.u[gg] = pw[((mi * 16 + lrow) * 64 + kk) >> 1];
        unsigned lo = vu[kk * 32 + ni * 16 + lrow];         // V^T fragment (strided)
        unsigned hi = vu[(kk + 1) * 32 + ni * 16 + lrow];
        bf.u[gg] = lo | (hi << 16);
      }
      acc = __builtin_amdgcn_wmma_f32_16x16x32_bf16(false, af.v, false, bf.v, (short)0, acc, false, false);
    }
    int d = ni * 16 + lrow;
#pragma unroll
    for (int r = 0; r < 8; ++r) {
      int i = mi * 16 + r + 8 * half;
      out[((size_t)(win * 64 + i)) * CB + h * 32 + d] = f2bf(acc[r]);
    }
  }
}

// ---------------------------------------------------------------- depthwise 3x3 + BN + ReLU
__global__ __launch_bounds__(256) void dwconv_kernel(const __bf16* __restrict__ in,
                                                     const float* __restrict__ wgt,
                                                     const float* __restrict__ bng,
                                                     const float* __restrict__ bnb,
                                                     __bf16* __restrict__ out) {
  size_t idx = (size_t)blockIdx.x * 256 + threadIdx.x;  // B*H*W*HID
  int c = (int)(idx & 1023);
  size_t tok = idx >> 10;
  int bb = (int)(tok >> 14); int rem = (int)(tok & 16383);
  int y = rem >> 7, x = rem & 127;
  float acc = 0.f;
#pragma unroll
  for (int dy = -1; dy <= 1; ++dy)
#pragma unroll
    for (int dx = -1; dx <= 1; ++dx) {
      int yy = y + dy, xx = x + dx;
      if (yy >= 0 && yy < 128 && xx >= 0 && xx < 128) {
        float v = bf2f(in[(((size_t)bb * 16384 + yy * 128 + xx) << 10) + c]);
        acc += v * wgt[c * 9 + (dy + 1) * 3 + (dx + 1)];
      }
    }
  acc *= 0.9999950000374997f;                  // rsqrt(1 + 1e-5)
  acc = acc * bng[c] + bnb[c];
  out[idx] = f2bf(fmaxf(acc, 0.f));
}

// ---------------------------------------------------------------- launch
extern "C" void kernel_launch(void* const* d_in, const int* in_sizes, int n_in,
                              void* d_out, int out_size, void* d_ws, size_t ws_size,
                              hipStream_t stream) {
  const float* x      = (const float*)d_in[0];
  const float* g1     = (const float*)d_in[3];
  const float* b1     = (const float*)d_in[4];
  const float* qkv_w  = (const float*)d_in[5];
  const float* qkv_b  = (const float*)d_in[6];
  const float* rpb    = (const float*)d_in[7];
  const int*   rel    = (const int*)d_in[8];
  const float* proj_w = (const float*)d_in[9];
  const float* g2     = (const float*)d_in[10];
  const float* b2     = (const float*)d_in[11];
  const float* up_w   = (const float*)d_in[12];
  const float* up_b   = (const float*)d_in[13];
  const float* dw_w   = (const float*)d_in[14];
  const float* bn_g   = (const float*)d_in[15];
  const float* bn_b   = (const float*)d_in[16];
  const float* down_w = (const float*)d_in[17];
  const float* down_b = (const float*)d_in[18];

  char* ws = (char*)d_ws;
  __bf16* tokbuf = (__bf16*)(ws);                           // 33.5 MB (reused 3x)
  __bf16* qkvbuf = (__bf16*)(ws + (size_t)33554432);        // 100.7 MB
  float*  x1     = (float*) (ws + (size_t)134217728);       // 67 MB
  float*  scbuf  = (float*) (ws + (size_t)201326592);       // 67 MB
  __bf16* hid    = (__bf16*)(ws + (size_t)268435456);       // 134 MB
  __bf16* dwout  = (__bf16*)(ws + (size_t)402653184);       // 134 MB
  __bf16* wq     = (__bf16*)(ws + (size_t)536870912);
  __bf16* wp     = (__bf16*)(ws + (size_t)537264128);
  __bf16* wu     = (__bf16*)(ws + (size_t)537395200);
  __bf16* wd     = (__bf16*)(ws + (size_t)537919488);

  // weights -> bf16
  cvt_bf16_kernel<<<(768 * 256 + 255) / 256, 256, 0, stream>>>(qkv_w, wq, 768 * 256);
  cvt_bf16_kernel<<<(256 * 256 + 255) / 256, 256, 0, stream>>>(proj_w, wp, 256 * 256);
  cvt_bf16_kernel<<<(1024 * 256 + 255) / 256, 256, 0, stream>>>(up_w, wu, 1024 * 256);
  cvt_bf16_kernel<<<(256 * 1024 + 255) / 256, 256, 0, stream>>>(down_w, wd, 256 * 1024);

  // LN1 + window partition (bf16)
  ln_kernel<<<NTOK / 8, 256, 0, stream>>>(x, g1, b1, tokbuf, nullptr, 1);

  // QKV GEMM: 65536x768x256
  gemm_bf16_kernel<<<dim3(12, 512), 256, 0, stream>>>(
      tokbuf, 256, wq, qkv_b, NTOK, 768, 256, 0, qkvbuf, 768, nullptr, nullptr, nullptr);

  // windowed attention (writes back into tokbuf)
  attn_kernel<<<NWIN * 8, 256, 0, stream>>>(qkvbuf, rpb, rel, tokbuf);

  // proj GEMM + un-permute + residual -> x1
  gemm_bf16_kernel<<<dim3(4, 512), 256, 0, stream>>>(
      tokbuf, 256, wp, nullptr, NTOK, 256, 256, 1, nullptr, 0, x1, x, nullptr);

  // LN2 -> sc (f32) + bf16 activation (tokbuf reused)
  ln_kernel<<<NTOK / 8, 256, 0, stream>>>(x1, g2, b2, tokbuf, scbuf, 0);

  // up GEMM + ReLU: 65536x1024x256
  gemm_bf16_kernel<<<dim3(16, 512), 256, 0, stream>>>(
      tokbuf, 256, wu, up_b, NTOK, 1024, 256, 2, hid, 1024, nullptr, nullptr, nullptr);

  // depthwise 3x3 + BN + ReLU
  dwconv_kernel<<<(NTOK * 1024) / 256, 256, 0, stream>>>(hid, dw_w, bn_g, bn_b, dwout);

  // down GEMM + bias + sc + x1 -> out : 65536x256x1024
  gemm_bf16_kernel<<<dim3(4, 512), 256, 0, stream>>>(
      dwout, 1024, wd, down_b, NTOK, 256, 1024, 3, nullptr, 0, (float*)d_out, x1, scbuf);
}
